// SelfAttnTDM_56564719288659
// MI455X (gfx1250) — compile-verified, matrix operand order checked by reading, not run on hardware
//
#include <hip/hip_runtime.h>
#include <hip/hip_bf16.h>
#include <math.h>

typedef __attribute__((ext_vector_type(16))) _Float16 v16h;
typedef __attribute__((ext_vector_type(8)))  _Float16 v8h;
typedef __attribute__((ext_vector_type(8)))  float    v8f;

#define GEMM_ASYNC 1  // stage W tiles to LDS with global_load_async_to_lds_b128

__device__ __forceinline__ v16h pack16(v8h a, v8h b) {
  v16h r;
#pragma unroll
  for (int i = 0; i < 8; i++) { r[i] = a[i]; r[8 + i] = b[i]; }
  return r;
}

__device__ __forceinline__ void wait_asynccnt0() {
#if defined(__has_builtin) && __has_builtin(__builtin_amdgcn_s_wait_asynccnt)
  __builtin_amdgcn_s_wait_asynccnt(0);
#else
  asm volatile("s_wait_asynccnt 0x0" ::: "memory");
#endif
}

// ---------------------------------------------------------------------------
// f32 -> f16 conversion (weights)
// ---------------------------------------------------------------------------
__global__ void cvt_kernel(const float* __restrict__ in, _Float16* __restrict__ out, int n) {
  int i = blockIdx.x * 256 + threadIdx.x;
  if (i < n) out[i] = (_Float16)in[i];
}

// ---------------------------------------------------------------------------
// x[B,C,T] -> out0[T,B,C] = x*sqrt(C) + PE ; also f16 copy for stage 0
// ---------------------------------------------------------------------------
__global__ void prep_kernel(const float* __restrict__ x, float* __restrict__ out0,
                            _Float16* __restrict__ xh, int total) {
  int idx = blockIdx.x * 256 + threadIdx.x;
  if (idx >= total) return;
  int c = idx & 2047;
  int b = (idx >> 11) & 1;
  int t = idx >> 12;
  float v = x[((size_t)(b * 2048 + c)) * 4096 + t];
  int i2 = c & ~1;
  float freq = __expf((float)i2 * (-9.210340371976184f / 2048.f));
  float ang = (float)t * freq;
  float pe = (c & 1) ? cosf(ang) : sinf(ang);
  float y = v * 45.254833995939045f + pe;  // sqrt(2048)
  out0[idx] = y;
  xh[idx] = (_Float16)y;
}

// ---------------------------------------------------------------------------
// LayerNorm over 256 cols; one wave32 per row; writes f32 + f16 copies
// ---------------------------------------------------------------------------
__global__ __launch_bounds__(256) void ln_kernel(const float* __restrict__ in,
                                                 const float* __restrict__ g,
                                                 const float* __restrict__ b,
                                                 float* __restrict__ outF,
                                                 _Float16* __restrict__ outH, int M) {
  int row = blockIdx.x * 8 + (threadIdx.x >> 5);
  if (row >= M) return;
  int lane = threadIdx.x & 31;
  const float* p = in + (size_t)row * 256 + lane * 8;
  float v[8];
#pragma unroll
  for (int i = 0; i < 8; i++) v[i] = p[i];
  float s = 0.f, s2 = 0.f;
#pragma unroll
  for (int i = 0; i < 8; i++) { s += v[i]; s2 += v[i] * v[i]; }
#pragma unroll
  for (int m = 1; m <= 16; m <<= 1) { s += __shfl_xor(s, m); s2 += __shfl_xor(s2, m); }
  float mu = s * (1.f / 256.f);
  float var = s2 * (1.f / 256.f) - mu * mu;
  float rstd = rsqrtf(var + 1e-5f);
#pragma unroll
  for (int i = 0; i < 8; i++) {
    int c = lane * 8 + i;
    float y = (v[i] - mu) * rstd * g[c] + b[c];
    outF[(size_t)row * 256 + c] = y;
    outH[(size_t)row * 256 + c] = (_Float16)y;
  }
}

// ---------------------------------------------------------------------------
// Generic WMMA GEMM: Y[M,N] = A[M,K](f16) * W[N,K]^T(f16)  (+bias)(+relu)(+resid)
// DS mode fuses downsample: logical A row r = tn*2+b is concat of source rows
// (2tn, b) and (2tn+1, b) of width prevC (=K/2).
// Wave computes 16M x 64N; block = 8 waves = 32M x 256N tile.
// W tile (256 x 32 f16 per K-step) is staged into LDS via
// GLOBAL_LOAD_ASYNC_TO_LDS_B128, double-buffered with ASYNCcnt + barriers.
// LDS row stride padded to 40 halves (80B) to spread B-frag reads over banks.
// ---------------------------------------------------------------------------
template <int DS, int RELU>
__global__ __launch_bounds__(256) void gemm_kernel(
    const _Float16* __restrict__ A, const _Float16* __restrict__ W,
    const float* __restrict__ bias, const float* __restrict__ resid,
    float* __restrict__ outF, _Float16* __restrict__ outH,
    int M, int N, int K, int prevC) {
  __shared__ alignas(16) _Float16 wtile[2][256][40];

  const int tid = threadIdx.x;
  const int w = tid >> 5;
  const int lane = tid & 31;
  const int half = lane >> 4;
  const int l16 = lane & 15;
  const int mbase = blockIdx.x * 32 + (w >> 2) * 16;
  const int nlocal = (w & 3) * 64;                // wave's N offset inside block tile
  const int nblock = blockIdx.y * 256;            // block's N base

  // Per-lane A row source pointers
  const int r = mbase + l16;
  const _Float16 *arow0, *arow1;
  if (DS) {
    int tn = r >> 1, bb = r & 1;
    arow0 = A + (size_t)((tn * 2 + 0) * 2 + bb) * prevC;
    arow1 = A + (size_t)((tn * 2 + 1) * 2 + bb) * prevC;
  } else {
    arow0 = A + (size_t)r * K;
    arow1 = arow0;
  }

  // Cooperative W staging: thread t copies row (nblock + t), 32 f16 per K-step.
  const _Float16* wsrc_row = W + (size_t)(nblock + tid) * K;
  auto fill = [&](int buf, int ks) {
    const _Float16* src = wsrc_row + ks * 32;
#if GEMM_ASYNC
    unsigned ldsa = (unsigned)(size_t)&wtile[buf][tid][0];
#pragma unroll
    for (int c = 0; c < 4; c++) {
      asm volatile("global_load_async_to_lds_b128 %0, %1, off"
                   :: "v"(ldsa + c * 16), "v"(src + c * 8)
                   : "memory");
    }
#else
#pragma unroll
    for (int c = 0; c < 4; c++)
      *(v8h*)&wtile[buf][tid][c * 8] = *(const v8h*)(src + c * 8);
#endif
  };

  v8f acc[4];
#pragma unroll
  for (int f = 0; f < 4; f++)
#pragma unroll
    for (int v = 0; v < 8; v++) acc[f][v] = 0.f;

  const int ksteps = K >> 5;
  fill(0, 0);
  for (int ks = 0; ks < ksteps; ks++) {
    int buf = ks & 1;
#if GEMM_ASYNC
    wait_asynccnt0();              // my async fills for 'buf' landed in LDS
#endif
    __syncthreads();               // everyone's fills for 'buf' complete
    if (ks + 1 < ksteps) fill(buf ^ 1, ks + 1);

    int k0 = ks * 32 + half * 8;   // chunk g=0 : k = half*8 + p
    int k1 = k0 + 16;              // chunk g=1 : k = 16 + half*8 + p
    v8h a0, a1;
    if (DS) {
      a0 = *(const v8h*)((k0 < prevC) ? (arow0 + k0) : (arow1 + (k0 - prevC)));
      a1 = *(const v8h*)((k1 < prevC) ? (arow0 + k1) : (arow1 + (k1 - prevC)));
    } else {
      a0 = *(const v8h*)(arow0 + k0);
      a1 = *(const v8h*)(arow0 + k1);
      if (ks + 2 < ksteps) __builtin_prefetch(arow0 + k0 + 64, 0, 1);  // global_prefetch_b8
    }
    v16h af = pack16(a0, a1);
#pragma unroll
    for (int f = 0; f < 4; f++) {
      const _Float16* wr = &wtile[buf][nlocal + f * 16 + l16][0];
      v16h bf = pack16(*(const v8h*)(wr + half * 8), *(const v8h*)(wr + 16 + half * 8));
      acc[f] = __builtin_amdgcn_wmma_f32_16x16x32_f16(
          false, af, false, bf, (short)0, acc[f], false, false);
    }
    __syncthreads();               // all waves done reading 'buf' before refill
  }

#pragma unroll
  for (int f = 0; f < 4; f++) {
    int col = nblock + nlocal + f * 16 + l16;
    float bv = bias ? bias[col] : 0.f;
#pragma unroll
    for (int v = 0; v < 8; v++) {
      int rr = mbase + v + 8 * half;  // D layout: lanes16-31 hold M=8..15
      float x = acc[f][v] + bv;
      if (RELU) x = fmaxf(x, 0.f);
      size_t idx = (size_t)rr * N + col;
      if (resid) x += resid[idx];
      if (outF) outF[idx] = x;
      if (outH) outH[idx] = (_Float16)x;
    }
  }
}

// ---------------------------------------------------------------------------
// Banded attention. One wave per (b, h, query-tile of 16). Keys restricted to
// the 64-token window [ (qt-1)*16, (qt+3)*16 ) which covers the +/-16 band.
// S = Q K^T via 4 WMMAs; softmax via shfl_xor within 16-lane row groups;
// P staged through LDS into A-fragment layout; out = P V via 4 WMMAs.
// qkvh rows: token r = t*2+b, cols [0:256)=Q [256:512)=K [512:768)=V.
// ---------------------------------------------------------------------------
__global__ __launch_bounds__(128) void attn_kernel(const _Float16* __restrict__ qkvh,
                                                   _Float16* __restrict__ aoh, int T) {
  __shared__ alignas(16) _Float16 plds[4][16][64];
  const int w = threadIdx.x >> 5;
  const int lane = threadIdx.x & 31;
  const int half = lane >> 4;
  const int l16 = lane & 15;
  const int ntq = T >> 4;
  int gid = blockIdx.x * 4 + w;
  int qt = gid % ntq;
  int bh = gid / ntq;
  int b = bh & 1;
  int h = bh >> 1;
  const float scale = 0.17677669529663687f;  // 1/sqrt(32)

  // Q fragment (A layout): lane row m = l16, dims k = g*16 + half*8 + p
  int tq = qt * 16 + l16;
  const _Float16* qrow = qkvh + (size_t)(tq * 2 + b) * 768 + h * 32;
  v16h qf = pack16(*(const v8h*)(qrow + half * 8), *(const v8h*)(qrow + 16 + half * 8));

  v8f s[4];
#pragma unroll
  for (int tt = 0; tt < 4; tt++)
#pragma unroll
    for (int v = 0; v < 8; v++) s[tt][v] = 0.f;

#pragma unroll
  for (int tt = 0; tt < 4; tt++) {
    int ktile = qt - 1 + tt;
    v16h kf;
    if ((unsigned)ktile < (unsigned)ntq) {
      int tk = ktile * 16 + l16;
      const _Float16* krow = qkvh + (size_t)(tk * 2 + b) * 768 + 256 + h * 32;
      kf = pack16(*(const v8h*)(krow + half * 8), *(const v8h*)(krow + 16 + half * 8));
    } else {
#pragma unroll
      for (int i = 0; i < 16; i++) kf[i] = (_Float16)0.f;
    }
    s[tt] = __builtin_amdgcn_wmma_f32_16x16x32_f16(
        false, qf, false, kf, (short)0, s[tt], false, false);
  }

  // scale + band mask + softmax (row m = v + 8*half, col j per tile/lane)
#pragma unroll
  for (int v = 0; v < 8; v++) {
    int i = qt * 16 + v + 8 * half;
    float rm = -3.4e38f;
#pragma unroll
    for (int tt = 0; tt < 4; tt++) {
      int j = (qt - 1 + tt) * 16 + l16;
      float x = s[tt][v] * scale;
      int d = i - j;
      if (d < 0) d = -d;
      if (d > 16 || j < 0 || j >= T) x = -1e9f;
      s[tt][v] = x;
      rm = fmaxf(rm, x);
    }
#pragma unroll
    for (int m = 1; m <= 8; m <<= 1) rm = fmaxf(rm, __shfl_xor(rm, m));
    float rs = 0.f;
#pragma unroll
    for (int tt = 0; tt < 4; tt++) {
      float p = __expf(s[tt][v] - rm);
      s[tt][v] = p;
      rs += p;
    }
#pragma unroll
    for (int m = 1; m <= 8; m <<= 1) rs += __shfl_xor(rs, m);
    float inv = 1.f / rs;
#pragma unroll
    for (int tt = 0; tt < 4; tt++)
      plds[w][v + 8 * half][tt * 16 + l16] = (_Float16)(s[tt][v] * inv);
  }
  __syncthreads();

  v8f o[2];
#pragma unroll
  for (int nt = 0; nt < 2; nt++)
#pragma unroll
    for (int v = 0; v < 8; v++) o[nt][v] = 0.f;

#pragma unroll
  for (int ss = 0; ss < 2; ss++) {
    const _Float16* pr = &plds[w][l16][0];
    v16h pf = pack16(*(const v8h*)(pr + ss * 32 + half * 8),
                     *(const v8h*)(pr + ss * 32 + 16 + half * 8));
#pragma unroll
    for (int nt = 0; nt < 2; nt++) {
      v16h vf;  // B frag of V: lane col n = dim, k = key index
#pragma unroll
      for (int j = 0; j < 16; j++) {
        int g = j >> 3, p = j & 7;
        int kk = ss * 32 + g * 16 + half * 8 + p;
        int tk = (qt - 1) * 16 + kk;
        _Float16 val = (_Float16)0.f;
        if (tk >= 0 && tk < T)
          val = qkvh[(size_t)(tk * 2 + b) * 768 + 512 + h * 32 + nt * 16 + l16];
        vf[j] = val;
      }
      o[nt] = __builtin_amdgcn_wmma_f32_16x16x32_f16(
          false, pf, false, vf, (short)0, o[nt], false, false);
    }
  }

#pragma unroll
  for (int nt = 0; nt < 2; nt++)
#pragma unroll
    for (int v = 0; v < 8; v++) {
      int t = qt * 16 + v + 8 * half;
      aoh[(size_t)(t * 2 + b) * 256 + h * 32 + nt * 16 + l16] = (_Float16)o[nt][v];
    }
}

// ---------------------------------------------------------------------------
// Host orchestration
// ---------------------------------------------------------------------------
extern "C" void kernel_launch(void* const* d_in, const int* in_sizes, int n_in,
                              void* d_out, int out_size, void* d_ws, size_t ws_size,
                              hipStream_t stream) {
  (void)in_sizes; (void)n_in; (void)out_size; (void)ws_size;
  const float* x = (const float*)d_in[0];
  struct SP {
    const float *Wr, *ln_g, *ln_b, *Win, *bin, *Wout, *bout, *W1, *b1, *W2, *b2,
        *n1g, *n1b, *n2g, *n2b;
  };
  SP sp[4];
  for (int s = 0; s < 4; s++) {
    int base = 1 + s * 15;
    sp[s].Wr   = (const float*)d_in[base + 0];
    sp[s].ln_g = (const float*)d_in[base + 1];
    sp[s].ln_b = (const float*)d_in[base + 2];
    sp[s].Win  = (const float*)d_in[base + 3];
    sp[s].bin  = (const float*)d_in[base + 4];
    sp[s].Wout = (const float*)d_in[base + 5];
    sp[s].bout = (const float*)d_in[base + 6];
    sp[s].W1   = (const float*)d_in[base + 7];
    sp[s].b1   = (const float*)d_in[base + 8];
    sp[s].W2   = (const float*)d_in[base + 9];
    sp[s].b2   = (const float*)d_in[base + 10];
    sp[s].n1g  = (const float*)d_in[base + 11];
    sp[s].n1b  = (const float*)d_in[base + 12];
    sp[s].n2g  = (const float*)d_in[base + 13];
    sp[s].n2b  = (const float*)d_in[base + 14];
  }

  char* wsp = (char*)d_ws;
  size_t off = 0;
  auto alloc = [&](size_t bytes) -> void* {
    void* p = wsp + off;
    off = (off + bytes + 255) & ~(size_t)255;
    return p;
  };

  _Float16 *WrH[4], *WinH[4], *WoutH[4], *W1H[4], *W2H[4];
  for (int s = 0; s < 4; s++) {
    int inc = s ? 512 : 4096;
    WrH[s]   = (_Float16*)alloc((size_t)256 * inc * 2);
    WinH[s]  = (_Float16*)alloc((size_t)768 * 256 * 2);
    WoutH[s] = (_Float16*)alloc((size_t)256 * 256 * 2);
    W1H[s]   = (_Float16*)alloc((size_t)1024 * 256 * 2);
    W2H[s]   = (_Float16*)alloc((size_t)256 * 1024 * 2);
  }
  _Float16* xh   = (_Float16*)alloc((size_t)4096 * 2 * 2048 * 2);
  float*    tmp  = (float*)   alloc((size_t)4096 * 256 * 4);
  float*    xa   = (float*)   alloc((size_t)4096 * 256 * 4);
  _Float16* xah  = (_Float16*)alloc((size_t)4096 * 256 * 2);
  _Float16* qkvh = (_Float16*)alloc((size_t)4096 * 768 * 2);
  _Float16* aoh  = (_Float16*)alloc((size_t)4096 * 256 * 2);
  float*    xb   = (float*)   alloc((size_t)4096 * 256 * 4);
  _Float16* xbh  = (_Float16*)alloc((size_t)4096 * 256 * 2);
  _Float16* ffh  = (_Float16*)alloc((size_t)4096 * 1024 * 2);

  auto cvt = [&](const float* src, _Float16* dst, int n) {
    cvt_kernel<<<(n + 255) / 256, 256, 0, stream>>>(src, dst, n);
  };
  for (int s = 0; s < 4; s++) {
    int inc = s ? 512 : 4096;
    cvt(sp[s].Wr, WrH[s], 256 * inc);
    cvt(sp[s].Win, WinH[s], 768 * 256);
    cvt(sp[s].Wout, WoutH[s], 256 * 256);
    cvt(sp[s].W1, W1H[s], 1024 * 256);
    cvt(sp[s].W2, W2H[s], 256 * 1024);
  }

  int total0 = 4096 * 2 * 2048;
  prep_kernel<<<total0 / 256, 256, 0, stream>>>(x, (float*)d_out, xh, total0);

  float* outp = (float*)d_out;
  size_t ooff = (size_t)total0;
  for (int s = 0; s < 4; s++) {
    int Tn = 4096 >> (s + 1);
    int M = Tn * 2;
    int Kin = s ? 512 : 4096;
    // (a) downsample + Wr projection
    gemm_kernel<1, 0><<<dim3(M / 32, 1), 256, 0, stream>>>(
        xh, WrH[s], nullptr, nullptr, tmp, nullptr, M, 256, Kin, Kin / 2);
    // (b) LN
    ln_kernel<<<M / 8, 256, 0, stream>>>(tmp, sp[s].ln_g, sp[s].ln_b, xa, xah, M);
    // (c) QKV projection
    gemm_kernel<0, 0><<<dim3(M / 32, 3), 256, 0, stream>>>(
        xah, WinH[s], sp[s].bin, nullptr, nullptr, qkvh, M, 768, 256, 0);
    // (d) banded attention
    attn_kernel<<<(Tn / 16) * 16 / 4, 128, 0, stream>>>(qkvh, aoh, Tn);
    // (e) output projection + residual
    gemm_kernel<0, 0><<<dim3(M / 32, 1), 256, 0, stream>>>(
        aoh, WoutH[s], sp[s].bout, xa, tmp, nullptr, M, 256, 256, 0);
    // (f) LN1
    ln_kernel<<<M / 8, 256, 0, stream>>>(tmp, sp[s].n1g, sp[s].n1b, xb, xbh, M);
    // (g) FF1 + relu
    gemm_kernel<0, 1><<<dim3(M / 32, 4), 256, 0, stream>>>(
        xbh, W1H[s], sp[s].b1, nullptr, nullptr, ffh, M, 1024, 256, 0);
    // (h) FF2 + residual
    gemm_kernel<0, 0><<<dim3(M / 32, 1), 256, 0, stream>>>(
        ffh, W2H[s], sp[s].b2, xb, tmp, nullptr, M, 256, 1024, 0);
    // (i) LN2 -> stage output (f32 in d_out) + next stage input (f16 xh)
    ln_kernel<<<M / 8, 256, 0, stream>>>(tmp, sp[s].n2g, sp[s].n2b, outp + ooff, xh, M);
    ooff += (size_t)M * 256;
  }
}
